// CoAttention_35364760715904
// MI455X (gfx1250) — matrix operand order repaired and checked by reading, not compiled, not run on hardware
//
#include <hip/hip_runtime.h>
#include <hip/hip_bf16.h>

typedef __attribute__((ext_vector_type(16))) _Float16 v16h;
typedef __attribute__((ext_vector_type(8)))  _Float16 v8h;
typedef __attribute__((ext_vector_type(8)))  float    v8f;

#define HID   256
#define KDIM  768
#define BATCH 16
#define SEQ   1024
#define MROWS (BATCH * SEQ)   // 16384

#define WMMA_F16(A, B, C) \
  __builtin_amdgcn_wmma_f32_16x16x32_f16(false, (A), false, (B), (short)0, (C), false, false)

// ---- fragment loaders (CDNA5 wave32 WMMA layouts, CDNA5 ISA) ----

// A-matrix 16x32 f16: lane L holds row (L&15); halves 0..7 = K base+off+0..7,
// halves 8..15 = K base+off+16..23, off = (L>=16)?8:0.
static __device__ __forceinline__ v16h load_fragA_f16(const _Float16* p, int ld, int m0, int kb) {
  const int lane = threadIdx.x & 31;
  const _Float16* s = p + (size_t)(m0 + (lane & 15)) * ld + kb + ((lane >> 4) << 3);
  v8h lo = *(const v8h*)(s);
  v8h hi = *(const v8h*)(s + 16);
  v16h r;
#pragma unroll
  for (int i = 0; i < 8; ++i) { r[i] = lo[i]; r[i + 8] = hi[i]; }
  return r;
}

// Same A layout but converting from fp32 source on the fly.
static __device__ __forceinline__ v16h load_fragA_f32(const float* p, int ld, int m0, int kb) {
  const int lane = threadIdx.x & 31;
  const float* s = p + (size_t)(m0 + (lane & 15)) * ld + kb + ((lane >> 4) << 3);
  v16h r;
#pragma unroll
  for (int i = 0; i < 8; ++i) r[i] = (_Float16)s[i];
#pragma unroll
  for (int i = 0; i < 8; ++i) r[i + 8] = (_Float16)s[16 + i];
  return r;
}

// B-matrix 32x16 f16: lane L holds column (L&15); halves 0..7 = K kb+off+0..7,
// halves 8..15 = K kb+off+16..23, off = (L>=16)?8:0. Storage is row-major
// with "row = column-of-B" (i.e. B = storageT), so reads are contiguous.
static __device__ __forceinline__ v16h load_fragB(const _Float16* p, int ld, int n0, int kb) {
  const int lane = threadIdx.x & 31;
  const _Float16* s = p + (size_t)(n0 + (lane & 15)) * ld + kb + ((lane >> 4) << 3);
  v8h lo = *(const v8h*)(s);
  v8h hi = *(const v8h*)(s + 16);
  v16h r;
#pragma unroll
  for (int i = 0; i < 8; ++i) { r[i] = lo[i]; r[i + 8] = hi[i]; }
  return r;
}

// ---------------- kernel 1: convert weights fp32 -> f16 ----------------
__global__ void coattn_cvt_w(const float* __restrict__ Wq, const float* __restrict__ Wk,
                             const float* __restrict__ Wv, _Float16* __restrict__ Wh) {
  const int N = HID * KDIM;
  int i = blockIdx.x * blockDim.x + threadIdx.x;
  if (i < N) {
    Wh[i]         = (_Float16)Wq[i];
    Wh[N + i]     = (_Float16)Wk[i];
    Wh[2 * N + i] = (_Float16)Wv[i];
  }
}

// ---------------- kernel 2: q/k/v projections via WMMA ----------------
// grid = (MROWS/64, 2, 3), block = 128 (4 waves). Wave computes 16x128 tile.
// All 8 B-fragments of a K-chunk are preloaded before the WMMA chain so the
// scheduler can clause the 16 global_load_b128s and stage the loadcnt waits.
__global__ void coattn_proj(const float* __restrict__ x, const float* __restrict__ y,
                            const _Float16* __restrict__ Wh,
                            const float* __restrict__ bq, const float* __restrict__ bk,
                            const float* __restrict__ bv,
                            _Float16* __restrict__ qh, _Float16* __restrict__ kh,
                            _Float16* __restrict__ vT) {
  const int z    = blockIdx.z;
  const int wave = threadIdx.x >> 5;
  const int lane = threadIdx.x & 31;
  const int m0 = blockIdx.x * 64 + wave * 16;
  const int n0 = blockIdx.y * 128;

  const float*    X    = (z == 0) ? x : y;
  const _Float16* W    = Wh + (size_t)z * HID * KDIM;
  const float*    bias = (z == 0) ? bq : ((z == 1) ? bk : bv);

  v8f c[8];
#pragma unroll
  for (int nt = 0; nt < 8; ++nt) c[nt] = (v8f){0.f, 0.f, 0.f, 0.f, 0.f, 0.f, 0.f, 0.f};

  for (int kb = 0; kb < KDIM; kb += 32) {
    v16h a = load_fragA_f32(X, KDIM, m0, kb);
    v16h bfr[8];
#pragma unroll
    for (int nt = 0; nt < 8; ++nt)
      bfr[nt] = load_fragB(W, KDIM, n0 + nt * 16, kb);
#pragma unroll
    for (int nt = 0; nt < 8; ++nt)
      c[nt] = WMMA_F16(a, bfr[nt], c[nt]);
  }

  const int col  = lane & 15;
  const int roff = (lane >> 4) << 3;
#pragma unroll
  for (int nt = 0; nt < 8; ++nt) {
    const int n = n0 + nt * 16 + col;
    const float bn = bias[n];
#pragma unroll
    for (int r = 0; r < 8; ++r) {
      const int m = m0 + r + roff;
      const float val = c[nt][r] + bn;
      if (z == 0) {
        qh[(size_t)m * HID + n] = (_Float16)val;
      } else if (z == 1) {
        kh[(size_t)m * HID + n] = (_Float16)val;
      } else {
        const int b = m >> 10, t = m & 1023;
        vT[((size_t)b * HID + n) * SEQ + t] = (_Float16)val;
      }
    }
  }
}

// ---------------- kernel 3: fused flash attention ----------------
// grid = (SEQ/32, BATCH), block = 256 (8 waves).
// wave -> (qt = wave&1 : which 16-row q tile, hg = wave>>1 : 64-wide hidden slice).
// Per 32-key step: waves compute hidden-partial QK^T tiles -> LDS reduce ->
// 32 row-threads run online softmax -> P (f16) + alpha via LDS -> PV WMMA.
__global__ void coattn_attn(const _Float16* __restrict__ qh, const _Float16* __restrict__ kh,
                            const _Float16* __restrict__ vT, float* __restrict__ out) {
  __shared__ __align__(16) float    Spart[2][4][16][32];  // 16 KB partial scores
  __shared__ __align__(16) _Float16 Plds[2][16][32];      //  2 KB probs
  __shared__ float alphaS[32];
  __shared__ float linvS[32];

  const int tid  = threadIdx.x;
  const int lane = tid & 31;
  const int wave = tid >> 5;
  const int qt = wave & 1;
  const int hg = wave >> 1;

  const int b  = blockIdx.y;
  const int q0 = blockIdx.x * 32 + qt * 16;

  const _Float16* qb  = qh + (size_t)b * SEQ * HID;
  const _Float16* kbp = kh + (size_t)b * SEQ * HID;
  const _Float16* vtb = vT + (size_t)b * HID * SEQ;

  // q fragments for this wave's 64-wide hidden slice (2 K-chunks of 32)
  const v16h aq0 = load_fragA_f16(qb, HID, q0, hg * 64);
  const v16h aq1 = load_fragA_f16(qb, HID, q0, hg * 64 + 32);

  v8f oacc[4];
#pragma unroll
  for (int nt = 0; nt < 4; ++nt) oacc[nt] = (v8f){0.f, 0.f, 0.f, 0.f, 0.f, 0.f, 0.f, 0.f};

  float mrun = -1e30f, lrun = 0.f;  // meaningful for tid<32 (row threads)

  const int col  = lane & 15;
  const int roff = (lane >> 4) << 3;

  for (int it = 0; it < SEQ / 32; ++it) {
    const int key0 = it * 32;
    // prefetch next K/V tiles into cache hierarchy
    if (it + 1 < SEQ / 32) {
      __builtin_prefetch(kbp + (size_t)(key0 + 32 + col) * HID + hg * 64, 0, 1);
      __builtin_prefetch(vtb + (size_t)(hg * 64 + col) * SEQ + key0 + 32, 0, 1);
    }

    // ---- QK^T partials over this wave's hidden slice ----
    // Preload all 4 K-fragments, then run the WMMA chain (staged waits).
    {
      v16h kf[4];
#pragma unroll
      for (int kt = 0; kt < 2; ++kt) {
        kf[kt * 2 + 0] = load_fragB(kbp, HID, key0 + kt * 16, hg * 64);
        kf[kt * 2 + 1] = load_fragB(kbp, HID, key0 + kt * 16, hg * 64 + 32);
      }
      v8f s[2];
#pragma unroll
      for (int kt = 0; kt < 2; ++kt) {
        s[kt] = (v8f){0.f, 0.f, 0.f, 0.f, 0.f, 0.f, 0.f, 0.f};
        s[kt] = WMMA_F16(aq0, kf[kt * 2 + 0], s[kt]);
        s[kt] = WMMA_F16(aq1, kf[kt * 2 + 1], s[kt]);
      }
#pragma unroll
      for (int kt = 0; kt < 2; ++kt)
#pragma unroll
        for (int r = 0; r < 8; ++r)
          Spart[qt][hg][r + roff][kt * 16 + col] = s[kt][r];
    }
    __syncthreads();

    // ---- online softmax, one thread per query row ----
    if (tid < 32) {
      const int rqt = tid >> 4, row = tid & 15;
      float sv[32];
      float mcur = -1e30f;
#pragma unroll
      for (int j = 0; j < 32; ++j) {
        float t = (Spart[rqt][0][row][j] + Spart[rqt][1][row][j] +
                   Spart[rqt][2][row][j] + Spart[rqt][3][row][j]) * 0.0625f;  // 1/sqrt(256)
        sv[j] = t;
        mcur = fmaxf(mcur, t);
      }
      const float mnew  = fmaxf(mrun, mcur);
      const float alpha = __expf(mrun - mnew);
      float ls = 0.f;
#pragma unroll
      for (int j = 0; j < 32; ++j) {
        float p = __expf(sv[j] - mnew);
        ls += p;
        Plds[rqt][row][j] = (_Float16)p;
      }
      lrun = lrun * alpha + ls;
      mrun = mnew;
      alphaS[tid] = alpha;
    }
    __syncthreads();

    // ---- rescale running O, then PV accumulation ----
    {
      float al[8];
#pragma unroll
      for (int r = 0; r < 8; ++r) al[r] = alphaS[qt * 16 + r + roff];

      // Preload all 4 V-fragments first (one clause, staged waits).
      v16h bf[4];
#pragma unroll
      for (int nt = 0; nt < 4; ++nt)
        bf[nt] = load_fragB(vtb, SEQ, hg * 64 + nt * 16, key0);

      v16h pf;  // A-fragment of P (16x32) from LDS
      {
        const int row = lane & 15, off = (lane >> 4) << 3;
#pragma unroll
        for (int i = 0; i < 8; ++i) {
          pf[i]     = Plds[qt][row][off + i];
          pf[i + 8] = Plds[qt][row][off + 16 + i];
        }
      }
#pragma unroll
      for (int nt = 0; nt < 4; ++nt) {
#pragma unroll
        for (int r = 0; r < 8; ++r) oacc[nt][r] *= al[r];
        oacc[nt] = WMMA_F16(pf, bf[nt], oacc[nt]);
      }
    }
    __syncthreads();
  }

  if (tid < 32) linvS[tid] = 1.0f / lrun;
  __syncthreads();

#pragma unroll
  for (int nt = 0; nt < 4; ++nt) {
    const int h = hg * 64 + nt * 16 + col;
#pragma unroll
    for (int r = 0; r < 8; ++r) {
      const int row = q0 + r + roff;
      out[((size_t)b * SEQ + row) * HID + h] = oacc[nt][r] * linvS[qt * 16 + r + roff];
    }
  }
}

// ---------------- launch ----------------
extern "C" void kernel_launch(void* const* d_in, const int* in_sizes, int n_in,
                              void* d_out, int out_size, void* d_ws, size_t ws_size,
                              hipStream_t stream) {
  const float* x  = (const float*)d_in[0];
  const float* y  = (const float*)d_in[1];
  const float* Wq = (const float*)d_in[2];
  const float* bq = (const float*)d_in[3];
  const float* Wk = (const float*)d_in[4];
  const float* bk = (const float*)d_in[5];
  const float* Wv = (const float*)d_in[6];
  const float* bv = (const float*)d_in[7];
  float* out = (float*)d_out;

  // workspace layout (f16): Wh[3*256*768] | qh[16384*256] | kh[16384*256] | vT[16*256*1024]
  _Float16* Wh = (_Float16*)d_ws;
  _Float16* qh = Wh + (size_t)3 * HID * KDIM;
  _Float16* kh = qh + (size_t)MROWS * HID;
  _Float16* vT = kh + (size_t)MROWS * HID;
  // total: ~26.4 MB

  {
    const int N = HID * KDIM;
    coattn_cvt_w<<<(N + 255) / 256, 256, 0, stream>>>(Wq, Wk, Wv, Wh);
  }
  {
    dim3 grid(MROWS / 64, 2, 3);
    coattn_proj<<<grid, 128, 0, stream>>>(x, y, Wh, bq, bk, bv, qh, kh, vT);
  }
  {
    dim3 grid(SEQ / 32, BATCH);
    coattn_attn<<<grid, 256, 0, stream>>>(qh, kh, vT, out);
  }
}
